// MoEBlock_28802050687488
// MI455X (gfx1250) — matrix operand order repaired and checked by reading, not compiled, not run on hardware
//
#include <hip/hip_runtime.h>
#include <hip/hip_bf16.h>
#include <math.h>
#include <stdint.h>

// ---------------------------------------------------------------------------
// MoE block for MI455X (gfx1250), wave32 + WMMA bf16, async-LDS staging,
// double-buffered k-loop pipeline.
//
//   hs[t, e*F+f] = c[t,e] * silu(x Wg_e) * (x Wu_e)    (c=1 for shared e=8)
//   out = hs[T, 9*F] @ [w_down_proj ; w_shared_down]   (one dense GEMM)
// ---------------------------------------------------------------------------

#define T_TOK 2048
#define H_DIM 1024
#define F_DIM 1024
#define NEXP  8
#define NE_ALL 9
#define KTOT  (NE_ALL * F_DIM)   // 9216
#define LDK   40                 // k-stride (elems) of LDS tiles, 80B rows

typedef __bf16 bf16_t;
typedef __attribute__((ext_vector_type(16))) __bf16 v16bf;
typedef __attribute__((ext_vector_type(8)))  __bf16 v8bf;
typedef __attribute__((ext_vector_type(8)))  float  v8f;

#define WMMA_BF16(a, b, c) \
  __builtin_amdgcn_wmma_f32_16x16x32_bf16(false, (a), false, (b), (short)0, (c), false, false)

// ---- gfx1250 async global->LDS copy (ASYNCcnt-tracked, 16B per lane) -------
__device__ __forceinline__ void async_copy_b128(void* lds_dst, const void* src) {
  uint32_t loff = (uint32_t)(uintptr_t)lds_dst;
  asm volatile("global_load_async_to_lds_b128 %0, %1, off"
               :: "v"(loff), "v"(src) : "memory");
}
__device__ __forceinline__ void wait_async0() {
  asm volatile("s_wait_asynccnt 0" ::: "memory");
}

// ---- fragment loaders (ISA 7.12.2 layouts, wave32), 2x ds_load_b128 each ---
__device__ __forceinline__ v16bf frag_a_lds(const bf16_t* base) {
  const int lane = threadIdx.x & 31;
  const bf16_t* row = base + (lane & 15) * LDK + (lane >> 4) * 8;
  v8bf lo = *(const v8bf*)(row);
  v8bf hi = *(const v8bf*)(row + 16);
  return __builtin_shufflevector(lo, hi, 0, 1, 2, 3, 4, 5, 6, 7,
                                 8, 9, 10, 11, 12, 13, 14, 15);
}
__device__ __forceinline__ v16bf frag_b_ldsT(const bf16_t* baseT) {
  const int lane = threadIdx.x & 31;
  const bf16_t* col = baseT + (lane & 15) * LDK + (lane >> 4) * 8;
  v8bf lo = *(const v8bf*)(col);
  v8bf hi = *(const v8bf*)(col + 16);
  return __builtin_shufflevector(lo, hi, 0, 1, 2, 3, 4, 5, 6, 7,
                                 8, 9, 10, 11, 12, 13, 14, 15);
}

// ---- split weight staging: global->reg (issue early), reg->LDS (late) ------
// 32k x 128n tile, 256 threads: thread owns 2 n-cols x 8 k.
__device__ __forceinline__ void wload_128(float2 (&r)[8], const float* W,
                                          int kbase, int ldw, int n0) {
  const int tid = threadIdx.x;
  const int n = (tid & 63) * 2;
  const int kg = tid >> 6;  // 0..3
#pragma unroll
  for (int j = 0; j < 8; ++j)
    r[j] = *(const float2*)&W[(size_t)(kbase + kg * 8 + j) * ldw + n0 + n];
}
__device__ __forceinline__ void wstore_128(bf16_t* sBt, const float2 (&r)[8]) {
  const int tid = threadIdx.x;
  const int n = (tid & 63) * 2;
  const int kg = tid >> 6;
  v8bf c0, c1;
#pragma unroll
  for (int j = 0; j < 8; ++j) { c0[j] = (bf16_t)r[j].x; c1[j] = (bf16_t)r[j].y; }
  *(v8bf*)&sBt[(n + 0) * LDK + kg * 8] = c0;
  *(v8bf*)&sBt[(n + 1) * LDK + kg * 8] = c1;
}
// 32k x 256n tile, 256 threads: thread owns 2 n-cols x 16 k.
__device__ __forceinline__ void wload_256(float2 (&r)[16], const float* W,
                                          int kbase, int ldw, int n0) {
  const int tid = threadIdx.x;
  const int n = (tid & 127) * 2;
  const int kg = tid >> 7;  // 0..1
#pragma unroll
  for (int j = 0; j < 16; ++j)
    r[j] = *(const float2*)&W[(size_t)(kbase + kg * 16 + j) * ldw + n0 + n];
}
__device__ __forceinline__ void wstore_256(bf16_t* sBt, const float2 (&r)[16]) {
  const int tid = threadIdx.x;
  const int n = (tid & 127) * 2;
  const int kg = tid >> 7;
  v8bf a0, a1, b0, b1;
#pragma unroll
  for (int j = 0; j < 8; ++j) {
    a0[j] = (bf16_t)r[j].x;     b0[j] = (bf16_t)r[j].y;
    a1[j] = (bf16_t)r[8 + j].x; b1[j] = (bf16_t)r[8 + j].y;
  }
  *(v8bf*)&sBt[(n + 0) * LDK + kg * 16 + 0] = a0;
  *(v8bf*)&sBt[(n + 0) * LDK + kg * 16 + 8] = a1;
  *(v8bf*)&sBt[(n + 1) * LDK + kg * 16 + 0] = b0;
  *(v8bf*)&sBt[(n + 1) * LDK + kg * 16 + 8] = b1;
}

// ---- kernel 1: x fp32 -> bf16, plus aux_loss tail of d_out -----------------
__global__ __launch_bounds__(256) void cvt_x_kernel(const float* __restrict__ x,
                                                    bf16_t* __restrict__ xb,
                                                    float* __restrict__ out) {
  size_t i = ((size_t)blockIdx.x * 256 + threadIdx.x) * 8;
#pragma unroll
  for (int j = 0; j < 8; ++j) xb[i + j] = (bf16_t)x[i + j];
  if (blockIdx.x == 0 && threadIdx.x == 0)
    out[(size_t)T_TOK * H_DIM] = 0.0f;  // aux_loss = 0
}

// ---- kernel 2: router: logits -> top2 -> softmax -> dense combine[T,8] -----
__global__ __launch_bounds__(256) void router_kernel(const float* __restrict__ x,
                                                     const float* __restrict__ wg,
                                                     float* __restrict__ combine) {
  __shared__ float sX[256][33];
  const int tid = threadIdx.x;
  const int t0 = blockIdx.x * 256;

  float logit[NEXP];
#pragma unroll
  for (int e = 0; e < NEXP; ++e) logit[e] = 0.0f;

  for (int hb = 0; hb < H_DIM; hb += 32) {
    __syncthreads();
#pragma unroll
    for (int j = 0; j < 32; ++j)
      sX[tid][j] = x[(size_t)(t0 + tid) * H_DIM + hb + j];
    __syncthreads();
#pragma unroll
    for (int j = 0; j < 32; ++j) {
      float xv = sX[tid][j];
#pragma unroll
      for (int e = 0; e < NEXP; ++e)
        logit[e] = fmaf(xv, wg[(hb + j) * NEXP + e], logit[e]);  // uniform -> s_load
    }
  }

  int i1 = 0; float v1 = logit[0];
#pragma unroll
  for (int e = 1; e < NEXP; ++e)
    if (logit[e] > v1) { v1 = logit[e]; i1 = e; }
  int i2 = -1; float v2 = -INFINITY;
#pragma unroll
  for (int e = 0; e < NEXP; ++e)
    if (e != i1 && logit[e] > v2) { v2 = logit[e]; i2 = e; }

  float e2 = __expf(v2 - v1);                              // v1 >= v2
  float inv = __builtin_amdgcn_rcpf(1.0f + e2);            // v_rcp_f32
  float w1 = inv, w2 = e2 * inv;

  const int t = t0 + tid;
#pragma unroll
  for (int e = 0; e < NEXP; ++e)
    combine[t * NEXP + e] = (e == i1) ? w1 : ((e == i2) ? w2 : 0.0f);
}

// ---- kernel 3: hs[t, e*F+f] = c[t,e]*silu(g)*u -----------------------------
// block tile 128x128, 8 waves as 2(wm)x4(wn); wave tile 64x32 (4 Msub x 2 Nsub)
// per k-step/wave: 8 A-loads + 8 B-loads : 16 WMMA (x2 matrices sharing A).
__global__ __launch_bounds__(256) void moe_gate_up_kernel(
    const bf16_t* __restrict__ xb,
    const float* __restrict__ w_gate_proj, const float* __restrict__ w_up_proj,
    const float* __restrict__ w_shared_gate, const float* __restrict__ w_shared_up,
    const float* __restrict__ combine, bf16_t* __restrict__ hs) {
  const int t0  = blockIdx.x * 128;
  const int nf0 = blockIdx.y * 128;
  const int e   = blockIdx.z;
  const float* Wg = (e < NEXP) ? (w_gate_proj + (size_t)e * H_DIM * F_DIM) : w_shared_gate;
  const float* Wu = (e < NEXP) ? (w_up_proj   + (size_t)e * H_DIM * F_DIM) : w_shared_up;

  __shared__ __align__(16) bf16_t sA[2][128 * LDK];    // 128 m x 32 k
  __shared__ __align__(16) bf16_t sG[2][128 * LDK];    // transposed 128 n x 32 k
  __shared__ __align__(16) bf16_t sU[2][128 * LDK];

  const int tid = threadIdx.x;
  const int wv  = tid >> 5;
  const int wm  = wv & 1;    // M offset wm*64
  const int wn  = wv >> 1;   // N offset wn*32
  const int am  = tid >> 1;            // async A staging: row
  const int ak  = (tid & 1) * 16;      // 16-elem chunk

  v8f ag[4][2], au[4][2];
#pragma unroll
  for (int s = 0; s < 4; ++s)
#pragma unroll
    for (int ns = 0; ns < 2; ++ns)
#pragma unroll
      for (int v = 0; v < 8; ++v) { ag[s][ns][v] = 0.0f; au[s][ns][v] = 0.0f; }

  float2 rg[8], ru[8];
  // prologue: stage k-chunk 0 into buffer 0
  {
    const bf16_t* asrc = &xb[(size_t)(t0 + am) * H_DIM + ak];
    async_copy_b128(&sA[0][am * LDK + ak + 0], asrc + 0);
    async_copy_b128(&sA[0][am * LDK + ak + 8], asrc + 8);
    wload_128(rg, Wg, 0, F_DIM, nf0);
    wload_128(ru, Wu, 0, F_DIM, nf0);
    wstore_128(sG[0], rg);
    wstore_128(sU[0], ru);
    wait_async0();
    __syncthreads();
  }

  for (int kk = 0; kk < H_DIM; kk += 32) {
    const int cur = (kk >> 5) & 1, nxt = cur ^ 1;
    const bool more = (kk + 32) < H_DIM;
    if (more) {  // issue next chunk early: async A + weight loads to regs
      const bf16_t* asrc = &xb[(size_t)(t0 + am) * H_DIM + kk + 32 + ak];
      async_copy_b128(&sA[nxt][am * LDK + ak + 0], asrc + 0);
      async_copy_b128(&sA[nxt][am * LDK + ak + 8], asrc + 8);
      wload_128(rg, Wg, kk + 32, F_DIM, nf0);
      wload_128(ru, Wu, kk + 32, F_DIM, nf0);
    }

    v16bf a[4];
#pragma unroll
    for (int s = 0; s < 4; ++s)
      a[s] = frag_a_lds(&sA[cur][(wm * 64 + s * 16) * LDK]);
#pragma unroll
    for (int ns = 0; ns < 2; ++ns) {
      v16bf bg = frag_b_ldsT(&sG[cur][(wn * 32 + ns * 16) * LDK]);
#pragma unroll
      for (int s = 0; s < 4; ++s) ag[s][ns] = WMMA_BF16(a[s], bg, ag[s][ns]);
      v16bf bu = frag_b_ldsT(&sU[cur][(wn * 32 + ns * 16) * LDK]);
#pragma unroll
      for (int s = 0; s < 4; ++s) au[s][ns] = WMMA_BF16(a[s], bu, au[s][ns]);
    }

    if (more) { wstore_128(sG[nxt], rg); wstore_128(sU[nxt], ru); }
    wait_async0();
    __syncthreads();
  }

  // epilogue: D layout -> (m = v + 8*half, n = lane&15)
  const int lane = tid & 31;
  const int nl = lane & 15;
  const int mh = (lane >> 4) * 8;
#pragma unroll
  for (int s = 0; s < 4; ++s)
#pragma unroll
    for (int ns = 0; ns < 2; ++ns)
#pragma unroll
      for (int v = 0; v < 8; ++v) {
        int t = t0 + wm * 64 + s * 16 + mh + v;
        int f = nf0 + wn * 32 + ns * 16 + nl;
        float g = ag[s][ns][v];
        float u = au[s][ns][v];
        float sig = __builtin_amdgcn_rcpf(1.0f + __expf(-g));  // fast sigmoid
        float c = (e < NEXP) ? combine[t * NEXP + e] : 1.0f;
        hs[(size_t)t * KTOT + (size_t)e * F_DIM + f] = (bf16_t)(g * sig * u * c);
      }
}

// ---- kernel 4: out[T,H] = hs[T,9216] @ [w_down_proj ; w_shared_down] -------
// block tile 128x256, 8 waves as 2(wm)x4(wn); wave tile 64x64 (4x4 subtiles)
// per k-step/wave: 8 A-loads + 8 B-loads : 16 WMMA (4x reuse both operands).
__global__ __launch_bounds__(256) void moe_down_kernel(
    const bf16_t* __restrict__ hs, const float* __restrict__ w_down,
    const float* __restrict__ w_shared_down, float* __restrict__ out) {
  const int t0  = blockIdx.x * 128;
  const int nh0 = blockIdx.y * 256;

  __shared__ __align__(16) bf16_t sA[2][128 * LDK];
  __shared__ __align__(16) bf16_t sBt[2][256 * LDK];   // transposed 256 n x 32 k

  const int tid = threadIdx.x;
  const int wv  = tid >> 5;
  const int wm  = wv & 1;    // M offset wm*64
  const int wn  = wv >> 1;   // N offset wn*64
  const int am  = tid >> 1;
  const int ak  = (tid & 1) * 16;

  v8f acc[4][4];
#pragma unroll
  for (int s = 0; s < 4; ++s)
#pragma unroll
    for (int ns = 0; ns < 4; ++ns)
#pragma unroll
      for (int v = 0; v < 8; ++v) acc[s][ns][v] = 0.0f;

  float2 rb[16];
  {
    const bf16_t* asrc = &hs[(size_t)(t0 + am) * KTOT + ak];
    async_copy_b128(&sA[0][am * LDK + ak + 0], asrc + 0);
    async_copy_b128(&sA[0][am * LDK + ak + 8], asrc + 8);
    wload_256(rb, w_down, 0, H_DIM, nh0);
    wstore_256(sBt[0], rb);
    wait_async0();
    __syncthreads();
  }

  for (int kk = 0; kk < KTOT; kk += 32) {
    const int cur = (kk >> 5) & 1, nxt = cur ^ 1;
    const bool more = (kk + 32) < KTOT;
    if (more) {
      const bf16_t* asrc = &hs[(size_t)(t0 + am) * KTOT + kk + 32 + ak];
      async_copy_b128(&sA[nxt][am * LDK + ak + 0], asrc + 0);
      async_copy_b128(&sA[nxt][am * LDK + ak + 8], asrc + 8);
      const int kn = kk + 32;
      if (kn < NEXP * F_DIM) wload_256(rb, w_down, kn, H_DIM, nh0);
      else                   wload_256(rb, w_shared_down, kn - NEXP * F_DIM, H_DIM, nh0);
    }

    v16bf a[4];
#pragma unroll
    for (int s = 0; s < 4; ++s)
      a[s] = frag_a_lds(&sA[cur][(wm * 64 + s * 16) * LDK]);
#pragma unroll
    for (int ns = 0; ns < 4; ++ns) {
      v16bf b = frag_b_ldsT(&sBt[cur][(wn * 64 + ns * 16) * LDK]);
#pragma unroll
      for (int s = 0; s < 4; ++s) acc[s][ns] = WMMA_BF16(a[s], b, acc[s][ns]);
    }

    if (more) wstore_256(sBt[nxt], rb);
    wait_async0();
    __syncthreads();
  }

  const int lane = tid & 31;
  const int nl = lane & 15;
  const int mh = (lane >> 4) * 8;
#pragma unroll
  for (int s = 0; s < 4; ++s)
#pragma unroll
    for (int ns = 0; ns < 4; ++ns)
#pragma unroll
      for (int v = 0; v < 8; ++v) {
        int t = t0 + wm * 64 + s * 16 + mh + v;
        int h = nh0 + wn * 64 + ns * 16 + nl;
        out[(size_t)t * H_DIM + h] = acc[s][ns][v];  // exactly-once write
      }
}

// ---------------------------------------------------------------------------
extern "C" void kernel_launch(void* const* d_in, const int* in_sizes, int n_in,
                              void* d_out, int out_size, void* d_ws, size_t ws_size,
                              hipStream_t stream) {
  (void)in_sizes; (void)n_in; (void)out_size; (void)ws_size;

  const float* x              = (const float*)d_in[0];  // [1,2048,1024]
  const float* w_gate         = (const float*)d_in[1];  // [1024,8]
  const float* w_gate_proj    = (const float*)d_in[2];  // [8,1024,1024]
  const float* w_up_proj      = (const float*)d_in[3];  // [8,1024,1024]
  const float* w_down_proj    = (const float*)d_in[4];  // [8,1024,1024]
  const float* w_shared_gate  = (const float*)d_in[5];  // [1024,1024]
  const float* w_shared_up    = (const float*)d_in[6];  // [1024,1024]
  const float* w_shared_down  = (const float*)d_in[7];  // [1024,1024]
  float* out = (float*)d_out;                           // [2048*1024] + aux

  // workspace carve (needs ~41 MB): Xb @0 (4MB) | combine @4MB | hs @5MB (36MB)
  char* ws = (char*)d_ws;
  bf16_t* Xb      = (bf16_t*)ws;
  float*  combine = (float*)(ws + (size_t)(4u << 20));
  bf16_t* hs      = (bf16_t*)(ws + (size_t)(5u << 20));

  cvt_x_kernel<<<(T_TOK * H_DIM) / (256 * 8), 256, 0, stream>>>(x, Xb, out);
  router_kernel<<<T_TOK / 256, 256, 0, stream>>>(x, w_gate, combine);
  moe_gate_up_kernel<<<dim3(T_TOK / 128, F_DIM / 128, NE_ALL), 256, 0, stream>>>(
      Xb, w_gate_proj, w_up_proj, w_shared_gate, w_shared_up, combine, hs);
  moe_down_kernel<<<dim3(T_TOK / 128, H_DIM / 256), 256, 0, stream>>>(
      hs, w_down_proj, w_shared_down, out);
}
